// PatchContrastiveLoss_69312182223342
// MI455X (gfx1250) — matrix operand order repaired and checked by reading, not compiled, not run on hardware
//
#include <hip/hip_runtime.h>
#include <hip/hip_bf16.h>
#include <math.h>

typedef __attribute__((ext_vector_type(16))) __bf16 v16bf;
typedef __attribute__((ext_vector_type(8)))  __bf16 v8bf;
typedef __attribute__((ext_vector_type(8)))  float  v8f;
typedef __attribute__((ext_vector_type(4)))  int    v4i;

#define GLOBAL_AS __attribute__((address_space(1)))
#define SHARED_AS __attribute__((address_space(3)))

#define B_ROWS 4096
#define NROW   8192
#define DDIM   256
#define KCH    8          // 256 / 32
#define NBINS  512
#define KSEL   2047       // (2N-2)/4
#define CAP    1024
#define INV_T  14.285714285714286f
#define RANGE  14.30f
#define BIN_SCALE ((float)NBINS / (2.0f * RANGE))
#define RPW    64         // rows per workgroup (4 waves x 16)
#define BSTRIDE 264       // padded LDS stride (elems): 16B aligned, banks spread
#define NT     (NROW / 16)

__device__ __forceinline__ GLOBAL_AS v4i* as_global_v4i(const void* p) {
  return (GLOBAL_AS v4i*)(v4i*)const_cast<void*>(p);
}
__device__ __forceinline__ SHARED_AS v4i* as_shared_v4i(void* p) {
  return (SHARED_AS v4i*)(v4i*)p;
}

// ---------------- normalization + split + positives ----------------
__global__ __launch_bounds__(256) void prep_kernel(
    const float* __restrict__ ei, const float* __restrict__ ej,
    __bf16* __restrict__ ehi, __bf16* __restrict__ elo, float* __restrict__ pos) {
  int r = blockIdx.x, d = threadIdx.x;
  float xi = ei[r * DDIM + d], xj = ej[r * DDIM + d];
  __shared__ float s0[256], s1[256], s2[256];
  s0[d] = xi * xi; s1[d] = xj * xj; s2[d] = xi * xj;
  __syncthreads();
  for (int off = 128; off > 0; off >>= 1) {
    if (d < off) { s0[d] += s0[d + off]; s1[d] += s1[d + off]; s2[d] += s2[d + off]; }
    __syncthreads();
  }
  float rni = rsqrtf(fmaxf(s0[0], 1e-24f));
  float rnj = rsqrtf(fmaxf(s1[0], 1e-24f));
  float ni = xi * rni, nj = xj * rnj;
  __bf16 hi = (__bf16)ni;  __bf16 li = (__bf16)(ni - (float)hi);
  __bf16 hj = (__bf16)nj;  __bf16 lj = (__bf16)(nj - (float)hj);
  ehi[(size_t)r * DDIM + d] = hi;                    elo[(size_t)r * DDIM + d] = li;
  ehi[(size_t)(r + B_ROWS) * DDIM + d] = hj;         elo[(size_t)(r + B_ROWS) * DDIM + d] = lj;
  if (d == 0) { float p = s2[0] * rni * rnj * INV_T; pos[r] = p; pos[r + B_ROWS] = p; }
}

// ---------------- WMMA tile helpers ----------------
__device__ __forceinline__ void load_A(const __bf16* __restrict__ e, int rowbase, int lane,
                                       v16bf* __restrict__ A) {
  int m = lane & 15, half = lane >> 4;
  const __bf16* p = e + (size_t)(rowbase + m) * DDIM;
#pragma unroll
  for (int c = 0; c < KCH; c++) {
    v8bf x0 = *(const v8bf*)(p + c * 32 + half * 8);
    v8bf x1 = *(const v8bf*)(p + c * 32 + 16 + half * 8);
    v16bf a;
#pragma unroll
    for (int i = 0; i < 8; i++) { a[i] = x0[i]; a[8 + i] = x1[i]; }
    A[c] = a;
  }
}

// async-stage the 16-column B panel (full K, hi+lo) into LDS (128 threads, ASYNCcnt)
__device__ __forceinline__ void stage_B_async(const __bf16* __restrict__ ehi,
                                              const __bf16* __restrict__ elo,
                                              int colbase, int tid,
                                              __bf16* __restrict__ lbh, __bf16* __restrict__ lbl) {
#pragma unroll
  for (int i = 0; i < 4; i++) {
    int c2 = tid + i * 128;            // 512 chunks of 8 elems per matrix
    int col = c2 >> 5;                 // 32 chunks per column
    int ko  = (c2 & 31) * 8;
    const __bf16* gh = ehi + (size_t)(colbase + col) * DDIM + ko;
    const __bf16* gl = elo + (size_t)(colbase + col) * DDIM + ko;
    __bf16* sh = lbh + col * BSTRIDE + ko;
    __bf16* sl = lbl + col * BSTRIDE + ko;
    __builtin_amdgcn_global_load_async_to_lds_b128(as_global_v4i(gh), as_shared_v4i(sh), 0, 0);
    __builtin_amdgcn_global_load_async_to_lds_b128(as_global_v4i(gl), as_shared_v4i(sl), 0, 0);
  }
}

__device__ __forceinline__ void wait_async_barrier() {
  asm volatile("s_wait_asynccnt 0x0" ::: "memory");
  __syncthreads();
}

__device__ __forceinline__ v8f tile_mm_lds(const v16bf* __restrict__ Ah, const v16bf* __restrict__ Al,
                                           const __bf16* __restrict__ lbh, const __bf16* __restrict__ lbl,
                                           int lane) {
  int n = lane & 15, koff = (lane >> 4) * 8;
  const __bf16* ph = lbh + n * BSTRIDE + koff;
  const __bf16* pl = lbl + n * BSTRIDE + koff;
  v8f c = {};
#pragma unroll
  for (int k = 0; k < KCH; k++) {
    v8bf h0 = *(const v8bf*)(ph + k * 32);
    v8bf h1 = *(const v8bf*)(ph + k * 32 + 16);
    v8bf l0 = *(const v8bf*)(pl + k * 32);
    v8bf l1 = *(const v8bf*)(pl + k * 32 + 16);
    v16bf bh, bl;
#pragma unroll
    for (int i = 0; i < 8; i++) { bh[i] = h0[i]; bh[8 + i] = h1[i]; bl[i] = l0[i]; bl[8 + i] = l1[i]; }
    c = __builtin_amdgcn_wmma_f32_16x16x32_bf16(false, Ah[k], false, bh, (short)0, c, false, false);
    c = __builtin_amdgcn_wmma_f32_16x16x32_bf16(false, Ah[k], false, bl, (short)0, c, false, false);
    c = __builtin_amdgcn_wmma_f32_16x16x32_bf16(false, Al[k], false, bh, (short)0, c, false, false);
  }
  return c;
}

__device__ __forceinline__ int bin_of(float v) {
  int b = (int)((v + RANGE) * BIN_SCALE);
  return b < 0 ? 0 : (b > NBINS - 1 ? NBINS - 1 : b);
}

// ---------------- pass 1: histogram + sum exp(v), find critical bin ----------------
__global__ __launch_bounds__(128) void pass1_kernel(
    const __bf16* __restrict__ ehi, const __bf16* __restrict__ elo,
    int* __restrict__ crit, int* __restrict__ need, float* __restrict__ sumall) {
  __shared__ unsigned hcnt[RPW / 2][NBINS];        // 2 rows packed per u32 word: 64 KB
  __shared__ float rsum[RPW];
  __shared__ __bf16 lb[2][2][16 * BSTRIDE];        // [buf][hi/lo]: 33 KB
  int tid = threadIdx.x, lane = tid & 31, wave = tid >> 5;
  int rowbase = blockIdx.x * RPW + wave * 16;      // this wave's 16-row strip
  int lrow0 = wave * 16;
  for (int i = tid; i < (RPW / 2) * NBINS; i += 128) ((unsigned*)hcnt)[i] = 0u;
  if (tid < RPW) rsum[tid] = 0.f;

  v16bf Ah[KCH], Al[KCH];
  load_A(ehi, rowbase, lane, Ah);
  load_A(elo, rowbase, lane, Al);

  float accv[8];
#pragma unroll
  for (int r = 0; r < 8; r++) accv[r] = 0.f;
  int n = lane & 15, mh = (lane >> 4) * 8;

  stage_B_async(ehi, elo, 0, tid, lb[0][0], lb[0][1]);
  wait_async_barrier();

  for (int t = 0; t < NT; t++) {
    int cur = t & 1;
    if (t + 1 < NT)
      stage_B_async(ehi, elo, (t + 1) * 16, tid, lb[1 - cur][0], lb[1 - cur][1]);
    int colbase = t * 16;
    v8f c = tile_mm_lds(Ah, Al, lb[cur][0], lb[cur][1], lane);
    int gcol = colbase + n;
#pragma unroll
    for (int r = 0; r < 8; r++) {
      int grow = rowbase + mh + r;
      if (gcol == grow || gcol == ((grow + B_ROWS) & (NROW - 1))) continue;
      float v = c[r] * INV_T;
      int lr = lrow0 + mh + r;                      // lr & 1 == r & 1 (fold to constant)
      atomicAdd(&hcnt[lr >> 1][bin_of(v)], (r & 1) ? 0x10000u : 1u);
      accv[r] += __expf(v);
    }
    wait_async_barrier();
  }
#pragma unroll
  for (int r = 0; r < 8; r++) atomicAdd(&rsum[lrow0 + mh + r], accv[r]);
  __syncthreads();

  if (tid < RPW) {
    unsigned cum = 0; int cb = 0, nd = 0;
    int sh = (tid & 1) * 16;
    for (int b = NBINS - 1; b >= 0; b--) {
      unsigned cnb = (hcnt[tid >> 1][b] >> sh) & 0xFFFFu;
      if (cum + cnb >= (unsigned)KSEL) { cb = b; nd = KSEL - (int)cum; break; }
      cum += cnb;
    }
    int grow = blockIdx.x * RPW + tid;
    crit[grow] = cb;
    need[grow] = nd;
    sumall[grow] = rsum[tid];
  }
}

// ---------------- pass 2: sum exp(2v) above critical bin, compact critical bin ----------------
__global__ __launch_bounds__(128) void pass2_kernel(
    const __bf16* __restrict__ ehi, const __bf16* __restrict__ elo,
    const int* __restrict__ crit, float* __restrict__ topacc,
    float* __restrict__ lists, int* __restrict__ listcnt) {
  __shared__ int scrit[RPW];
  __shared__ float racc[RPW];
  __shared__ unsigned rcnt[RPW];
  __shared__ __bf16 lb[2][2][16 * BSTRIDE];
  int tid = threadIdx.x, lane = tid & 31, wave = tid >> 5;
  int rowbase = blockIdx.x * RPW + wave * 16;
  int lrow0 = wave * 16;
  if (tid < RPW) { scrit[tid] = crit[blockIdx.x * RPW + tid]; racc[tid] = 0.f; rcnt[tid] = 0u; }

  v16bf Ah[KCH], Al[KCH];
  load_A(ehi, rowbase, lane, Ah);
  load_A(elo, rowbase, lane, Al);

  float acc2[8];
#pragma unroll
  for (int r = 0; r < 8; r++) acc2[r] = 0.f;
  int n = lane & 15, mh = (lane >> 4) * 8;

  stage_B_async(ehi, elo, 0, tid, lb[0][0], lb[0][1]);
  wait_async_barrier();

  for (int t = 0; t < NT; t++) {
    int cur = t & 1;
    if (t + 1 < NT)
      stage_B_async(ehi, elo, (t + 1) * 16, tid, lb[1 - cur][0], lb[1 - cur][1]);
    int colbase = t * 16;
    v8f c = tile_mm_lds(Ah, Al, lb[cur][0], lb[cur][1], lane);
    int gcol = colbase + n;
#pragma unroll
    for (int r = 0; r < 8; r++) {
      int grow = rowbase + mh + r;
      if (gcol == grow || gcol == ((grow + B_ROWS) & (NROW - 1))) continue;
      float v = c[r] * INV_T;
      int b = bin_of(v), cb = scrit[lrow0 + mh + r];
      if (b > cb) {
        acc2[r] += __expf(2.f * v);
      } else if (b == cb) {
        unsigned idx = atomicAdd(&rcnt[lrow0 + mh + r], 1u);
        if (idx < CAP) lists[(size_t)grow * CAP + idx] = v;
      }
    }
    wait_async_barrier();
  }
#pragma unroll
  for (int r = 0; r < 8; r++) atomicAdd(&racc[lrow0 + mh + r], acc2[r]);
  __syncthreads();
  if (tid < RPW) {
    int grow = blockIdx.x * RPW + tid;
    topacc[grow] = racc[tid];
    unsigned cc = rcnt[tid];
    listcnt[grow] = (int)(cc > (unsigned)CAP ? (unsigned)CAP : cc);
  }
}

// ---------------- exact in-bin selection + per-row loss ----------------
__global__ __launch_bounds__(256) void select_kernel(
    const float* __restrict__ pos, const float* __restrict__ sumall,
    const float* __restrict__ topacc, const int* __restrict__ need,
    float* __restrict__ lists, const int* __restrict__ listcnt,
    float* __restrict__ loss) {
  int row = blockIdx.x * blockDim.x + threadIdx.x;
  if (row >= NROW) return;
  float* lst = lists + (size_t)row * CAP;
  int nn = listcnt[row];
  int r = need[row]; if (r > nn) r = nn;
  float stop = 0.f;
  for (int i = 0; i < r; i++) {           // partial selection sort (deterministic order)
    int mi = i; float mv = lst[i];
    for (int j = i + 1; j < nn; j++) { float x = lst[j]; if (x > mv) { mv = x; mi = j; } }
    lst[mi] = lst[i]; lst[i] = mv;
    stop += __expf(2.f * mv);
  }
  float p = pos[row];
  float total = __expf(p) + sumall[row] + topacc[row] + stop;
  loss[row] = -p + logf(total);
}

// ---------------- deterministic final reduction ----------------
__global__ __launch_bounds__(256) void final_kernel(const float* __restrict__ loss,
                                                    float* __restrict__ out) {
  __shared__ float s[256];
  int t = threadIdx.x;
  float a = 0.f;
  for (int i = t; i < NROW; i += 256) a += loss[i];
  s[t] = a; __syncthreads();
  for (int off = 128; off > 0; off >>= 1) { if (t < off) s[t] += s[t + off]; __syncthreads(); }
  if (t == 0) out[0] = s[0] / (float)NROW;
}

extern "C" void kernel_launch(void* const* d_in, const int* in_sizes, int n_in,
                              void* d_out, int out_size, void* d_ws, size_t ws_size,
                              hipStream_t stream) {
  (void)in_sizes; (void)n_in; (void)out_size; (void)ws_size;
  const float* ei = (const float*)d_in[0];
  const float* ej = (const float*)d_in[1];
  float* out = (float*)d_out;
  char* ws = (char*)d_ws;

  __bf16* ehi   = (__bf16*)ws;                        // 4 MB
  __bf16* elo   = (__bf16*)(ws + ((size_t)4 << 20));  // 4 MB
  float* pos    = (float*)(ws + ((size_t)8 << 20));   // 32 KB each below
  float* sumall = pos + NROW;
  int*   crit   = (int*)(sumall + NROW);
  int*   need   = crit + NROW;
  float* topacc = (float*)(need + NROW);
  int*   listcnt= (int*)(topacc + NROW);
  float* lossar = (float*)(listcnt + NROW);
  float* lists  = (float*)(ws + ((size_t)9 << 20));   // 32 MB

  prep_kernel  <<<B_ROWS,     256, 0, stream>>>(ei, ej, ehi, elo, pos);
  pass1_kernel <<<NROW / RPW, 128, 0, stream>>>(ehi, elo, crit, need, sumall);
  pass2_kernel <<<NROW / RPW, 128, 0, stream>>>(ehi, elo, crit, topacc, lists, listcnt);
  select_kernel<<<NROW / 256, 256, 0, stream>>>(pos, sumall, topacc, need, lists, listcnt, lossar);
  final_kernel <<<1,          256, 0, stream>>>(lossar, out);
}
